// CrossNetwork_91242285237034
// MI455X (gfx1250) — matrix operand order repaired and checked by reading, not compiled, not run on hardware
//
#include <hip/hip_runtime.h>
#include <hip/hip_bf16.h>
#include <math.h>

typedef __attribute__((ext_vector_type(2)))  float  v2f;
typedef __attribute__((ext_vector_type(8)))  float  v8f;
typedef __attribute__((ext_vector_type(8)))  __bf16 v8bf;
typedef __attribute__((ext_vector_type(16))) __bf16 v16bf;
typedef int v4i __attribute__((vector_size(16)));

#define B_DIM 16384
#define N_DIM 50
#define D_DIM 32
#define T_DIM 2
#define K_DIM 1600      // N*D
#define M_DIM 2500      // N*N
#define FEAT 3200       // T*N*D

// ---------------------------------------------------------------------------
// Async global->LDS copy (16B per lane), tracked by ASYNCcnt.
// Builtin signature (from hipcc diagnostic): param0 = int4 AS(1)*,
// so we pass (global int4*, lds int4*, imm offset, cpol).
// ---------------------------------------------------------------------------
#if defined(__has_builtin)
#if __has_builtin(__builtin_amdgcn_global_load_async_to_lds_b128)
#define HAVE_ASYNC_BUILTIN 1
#endif
#if __has_builtin(__builtin_amdgcn_s_wait_asynccnt)
#define HAVE_ASYNCCNT_BUILTIN 1
#endif
#endif

typedef __attribute__((address_space(1))) v4i* gp16_t;
typedef __attribute__((address_space(3))) v4i* lp16_t;

__device__ __forceinline__ void async_copy16(const void* gsrc, void* lds) {
#ifdef HAVE_ASYNC_BUILTIN
    __builtin_amdgcn_global_load_async_to_lds_b128(
        (gp16_t)gsrc, (lp16_t)lds, 0, 0);
#else
    const unsigned l = (unsigned)(uintptr_t)lds;
    asm volatile("global_load_async_to_lds_b128 %0, %1, off"
                 :: "v"(l), "v"((unsigned long long)(uintptr_t)gsrc)
                 : "memory");
#endif
}

template <int N>
__device__ __forceinline__ void wait_asynccnt() {
#ifdef HAVE_ASYNCCNT_BUILTIN
    __builtin_amdgcn_s_wait_asynccnt((unsigned short)N);
#else
    asm volatile("s_wait_asynccnt %0" :: "n"(N) : "memory");
#endif
}

// ---------------------------------------------------------------------------
// Kernel 0a: split f32 -> (hi bf16, lo bf16), RNE. hi+lo keeps ~16 mantissa
// bits so the split-bf16 GEMM (hi*hi + hi*lo + lo*hi) is fp32-grade.
// ---------------------------------------------------------------------------
__global__ __launch_bounds__(256) void cvt_split(
    const float* __restrict__ src, __bf16* __restrict__ hi,
    __bf16* __restrict__ lo, int n)
{
    int i = blockIdx.x * blockDim.x + threadIdx.x;
    const int stride = gridDim.x * blockDim.x;
    for (; i < n; i += stride) {
        const float f = src[i];
        const __bf16 h = (__bf16)f;
        const float r = f - (float)h;
        hi[i] = h;
        lo[i] = (__bf16)r;
    }
}

// ---------------------------------------------------------------------------
// Kernel 0b: transpose-pack masker_param (T,1600,2500) f32 into column-major
// hi/lo bf16 (T,2500,1600): wp[t][col][k]. Dword k-pairs then match the
// 16-bit WMMA B layout, so GEMM B-tiles are straight async 16B copies.
// ---------------------------------------------------------------------------
__global__ __launch_bounds__(256) void pack_w(
    const float* __restrict__ src, __bf16* __restrict__ hi,
    __bf16* __restrict__ lo)
{
    __shared__ float tile[32][33];
    const int t  = blockIdx.z;
    const int k0 = blockIdx.y * 32;
    const int c0 = blockIdx.x * 32;
    const int cc = threadIdx.x & 31;
    const int r0 = threadIdx.x >> 5;       // 0..7

    for (int r = r0; r < 32; r += 8) {
        float v = 0.f;
        if (c0 + cc < M_DIM)
            v = src[((size_t)t * K_DIM + k0 + r) * M_DIM + c0 + cc];
        tile[r][cc] = v;
    }
    __syncthreads();
    const int kk = threadIdx.x & 31;
    for (int cr = r0; cr < 32; cr += 8) {
        if (c0 + cr < M_DIM) {
            const float f = tile[kk][cr];
            const __bf16 h = (__bf16)f;
            const float rr = f - (float)h;
            const size_t o = ((size_t)t * M_DIM + c0 + cr) * K_DIM + k0 + kk;
            hi[o] = h;
            lo[o] = (__bf16)rr;
        }
    }
}

// ---------------------------------------------------------------------------
// Kernel 1: adj_raw[b,t,m] = relu( sum_k fusion[b,k]*W[t,k,m] + bias[t,m] )
// Split-bf16 WMMA GEMM, async-DMA double-buffered LDS pipeline.
// Block 256 = 8 waves; tile 128x128; wave owns a 128x16 strip (8 x v8f acc).
//
// LDS: 2 parities x 4 planes (Ahi,Alo,Bhi,Blo) x 128x40 bf16 = 80 KB.
// Stride 20 dwords => {20*r mod 64} permutes the 4-aligned banks, and every
// fragment / DMA chunk is an aligned 16B access.
//   A-frag (16-bit 16x32 ISA layout): lane<16 -> dwords {0..3, 8..11};
//                                     lane>=16 -> {4..7, 12..15}.
//   B-frag: same k order as A-frag (lane<16 -> {0..3, 8..11}; lane>=16 -> {4..7, 12..15}).
// Pipeline: issue tile i+2 after the consume barrier of tile i;
// s_wait_asynccnt 8 retires tile i while tile i+1 (8 ops/wave) is in flight.
// ---------------------------------------------------------------------------
#define SEG 5120   // 128*40 bf16 per plane

__global__ __launch_bounds__(256) void gemm_adj(
    const __bf16* __restrict__ xhi, const __bf16* __restrict__ xlo,   // (B,1600)
    const __bf16* __restrict__ wphi, const __bf16* __restrict__ wplo, // (T,2500,1600)
    const float*  __restrict__ bias,                                  // (T,2500)
    float* __restrict__ adj)                                          // (B,T,2500)
{
    __shared__ __bf16 s_tiles[2][4][SEG];

    const int t       = blockIdx.z;
    const int colBase = blockIdx.x * 128;
    const int rowBase = blockIdx.y * 128;
    const __bf16* __restrict__ Whi = wphi + (size_t)t * M_DIM * K_DIM;
    const __bf16* __restrict__ Wlo = wplo + (size_t)t * M_DIM * K_DIM;

    const int tid    = threadIdx.x;
    const int wave   = tid >> 5;
    const int lane   = tid & 31;
    const int laneLo = lane & 15;
    const int laneHi = lane >> 4;            // 0 or 1
    const int colLocal = wave * 16 + laneLo;
    const int col      = colBase + colLocal;

    v8f acc[8];
    #pragma unroll
    for (int mi = 0; mi < 8; ++mi)
        #pragma unroll
        for (int e = 0; e < 8; ++e) acc[mi][e] = 0.0f;

    // 8 async ops per wave per tile (2 chunks/thread x 4 planes), uniform:
    // OOB B columns are clamped (data duplicated, consumers store-guarded).
    auto issueTile = [&](int k0, int p) {
        __bf16* ah = s_tiles[p][0];
        __bf16* al = s_tiles[p][1];
        __bf16* bh = s_tiles[p][2];
        __bf16* bl = s_tiles[p][3];
        #pragma unroll
        for (int i = tid; i < 512; i += 256) {
            const int rc = i >> 2;           // row / local col, 0..127
            const int q  = i & 3;            // 16B chunk within 64B
            const size_t ga = (size_t)(rowBase + rc) * K_DIM + k0 + q * 8;
            async_copy16(xhi + ga, &ah[rc * 40 + q * 8]);
            async_copy16(xlo + ga, &al[rc * 40 + q * 8]);
            const int colc = (colBase + rc < M_DIM) ? (colBase + rc) : (M_DIM - 1);
            const size_t gb = (size_t)colc * K_DIM + k0 + q * 8;
            async_copy16(Whi + gb, &bh[rc * 40 + q * 8]);
            async_copy16(Wlo + gb, &bl[rc * 40 + q * 8]);
        }
    };

    issueTile(0, 0);
    issueTile(32, 1);

    for (int it = 0; it < 50; ++it) {
        if (it < 49) wait_asynccnt<8>();     // tile 'it' done; tile it+1 in flight
        else         wait_asynccnt<0>();
        __syncthreads();

        const int p = it & 1;
        const __bf16* ahp = s_tiles[p][0];
        const __bf16* alp = s_tiles[p][1];
        const __bf16* bhp = s_tiles[p][2];
        const __bf16* blp = s_tiles[p][3];

        const int bBase = colLocal * 40 + laneHi * 8;
        const v8bf bh0 = *(const v8bf*)(&bhp[bBase]);
        const v8bf bh1 = *(const v8bf*)(&bhp[bBase + 16]);
        const v8bf bl0 = *(const v8bf*)(&blp[bBase]);
        const v8bf bl1 = *(const v8bf*)(&blp[bBase + 16]);
        const v16bf Bhi = __builtin_shufflevector(bh0, bh1, 0,1,2,3,4,5,6,7,8,9,10,11,12,13,14,15);
        const v16bf Blo = __builtin_shufflevector(bl0, bl1, 0,1,2,3,4,5,6,7,8,9,10,11,12,13,14,15);

        #pragma unroll
        for (int mi = 0; mi < 8; ++mi) {
            const int aBase = (mi * 16 + laneLo) * 40 + laneHi * 8;
            const v8bf ah0 = *(const v8bf*)(&ahp[aBase]);
            const v8bf ah1 = *(const v8bf*)(&ahp[aBase + 16]);
            const v8bf al0 = *(const v8bf*)(&alp[aBase]);
            const v8bf al1 = *(const v8bf*)(&alp[aBase + 16]);
            const v16bf Ahi = __builtin_shufflevector(ah0, ah1, 0,1,2,3,4,5,6,7,8,9,10,11,12,13,14,15);
            const v16bf Alo = __builtin_shufflevector(al0, al1, 0,1,2,3,4,5,6,7,8,9,10,11,12,13,14,15);
            acc[mi] = __builtin_amdgcn_wmma_f32_16x16x32_bf16(
                false, Ahi, false, Bhi, (short)0, acc[mi], false, false);
            acc[mi] = __builtin_amdgcn_wmma_f32_16x16x32_bf16(
                false, Ahi, false, Blo, (short)0, acc[mi], false, false);
            acc[mi] = __builtin_amdgcn_wmma_f32_16x16x32_bf16(
                false, Alo, false, Bhi, (short)0, acc[mi], false, false);
        }
        __syncthreads();                     // everyone done reading parity p
        if (it + 2 < 50) issueTile((it + 2) * 32, p);
    }

    // --- epilogue: + bias, relu, store (C layout: vgpr v -> M=v or v+8) ---
    if (col < M_DIM) {
        const float bval = bias[t * M_DIM + col];
        const int rOff = laneHi ? 8 : 0;
        #pragma unroll
        for (int mi = 0; mi < 8; ++mi) {
            #pragma unroll
            for (int v = 0; v < 8; ++v) {
                const int grow = rowBase + mi * 16 + v + rOff;
                float val = acc[mi][v] + bval;
                val = val > 0.f ? val : 0.f;
                adj[(size_t)grow * (T_DIM * M_DIM) + t * M_DIM + col] = val;
            }
        }
    }
}

// ---------------------------------------------------------------------------
// Kernel 2: per (b,t): m = x@weight0; m2 = rownorm(adj)@m; GRU -> h
// Block: 128 threads = 4 waves; one wave per (b,t). lane == d (D=32=wave32).
// ---------------------------------------------------------------------------
__global__ __launch_bounds__(128) void msg_gru(
    const float* __restrict__ x,       // (B, 50, 32)
    const float* __restrict__ adj,     // (B, T, 2500) raw (un-normalized)
    const float* __restrict__ weight,  // (1, 32, 32)  [e][d]
    const float* __restrict__ w_ih,    // (96, 32)     [out][e]
    const float* __restrict__ w_hh,    // (96, 32)
    const float* __restrict__ b_ih,    // (96)
    const float* __restrict__ b_hh,    // (96)
    float* __restrict__ hbuf)          // (B, T, 50, 32)
{
    __shared__ float s_wih[32 * 96];   // transposed: [e][out]
    __shared__ float s_whh[32 * 96];
    __shared__ float s_wgt[32 * 32];   // [e][d]
    __shared__ float s_bih[96];
    __shared__ float s_bhh[96];
    __shared__ float s_m[4][N_DIM * 32];
    __shared__ float s_adj[4][64];

    const int tid  = threadIdx.x;
    const int wave = tid >> 5;
    const int lane = tid & 31;

    for (int i = tid; i < 96 * 32; i += 128) {
        const int out = i >> 5, e = i & 31;
        s_wih[e * 96 + out] = w_ih[i];
        s_whh[e * 96 + out] = w_hh[i];
    }
    for (int i = tid; i < 32 * 32; i += 128) s_wgt[i] = weight[i];
    if (tid < 96) { s_bih[tid] = b_ih[tid]; s_bhh[tid] = b_hh[tid]; }
    __syncthreads();

    const int pair = blockIdx.x * 4 + wave;   // 0 .. B*T-1
    const int b = pair >> 1;
    const int t = pair & 1;

    const float* __restrict__ xb   = x   + (size_t)b * K_DIM;
    const float* __restrict__ adjb = adj + (size_t)b * (T_DIM * M_DIM) + t * M_DIM;
    float* __restrict__ hout = hbuf + (size_t)b * FEAT + t * K_DIM;

    // m[n][d] = sum_e x[b,n,e] * weight[e][d]
    for (int n = 0; n < N_DIM; ++n) {
        const float xv = xb[n * 32 + lane];
        float mv = 0.f;
        #pragma unroll
        for (int e = 0; e < 32; ++e)
            mv = fmaf(__shfl(xv, e, 32), s_wgt[e * 32 + lane], mv);
        s_m[wave][n * 32 + lane] = mv;
    }
    __syncthreads();

    for (int n = 0; n < N_DIM; ++n) {
        const float a0 = adjb[n * N_DIM + lane];
        const float a1 = (lane < N_DIM - 32) ? adjb[n * N_DIM + 32 + lane] : 0.f;
        s_adj[wave][lane]      = a0;
        s_adj[wave][32 + lane] = a1;
        float rs = a0 + a1;
        #pragma unroll
        for (int off = 16; off > 0; off >>= 1) rs += __shfl_xor(rs, off, 32);
        const float scale = 1.f / (rs + 1e-6f);

        float m2 = 0.f;
        for (int j = 0; j < N_DIM; ++j)
            m2 = fmaf(s_adj[wave][j], s_m[wave][j * 32 + lane], m2);
        m2 *= scale;

        const float hv = xb[n * 32 + lane];

        float g[3];
        #pragma unroll
        for (int gi = 0; gi < 3; ++gi) {
            const int out = gi * 32 + lane;
            float acc = s_bih[out] + s_bhh[out];
            #pragma unroll
            for (int e = 0; e < 32; ++e) {
                acc = fmaf(__shfl(m2, e, 32), s_wih[e * 96 + out], acc);
                acc = fmaf(__shfl(hv, e, 32), s_whh[e * 96 + out], acc);
            }
            g[gi] = acc;
        }
        const float r  = 1.f / (1.f + __expf(-g[0]));
        const float z  = 1.f / (1.f + __expf(-g[1]));
        const float ng = tanhf(r * g[2]);
        hout[n * 32 + lane] = (1.f - z) * ng + z * hv;
    }
}

// ---------------------------------------------------------------------------
// Kernel 3a/3b: deterministic two-phase BatchNorm statistics over B.
// ---------------------------------------------------------------------------
__global__ __launch_bounds__(256) void bn_partial(
    const float* __restrict__ h, float* __restrict__ psum, float* __restrict__ psumsq)
{
    const int c = blockIdx.x * 256 + threadIdx.x;
    if (c >= FEAT) return;
    const int chunk = blockIdx.y;           // 64 chunks of 256 rows
    const float* p = h + (size_t)chunk * 256 * FEAT + c;
    float s = 0.f, ss = 0.f;
    for (int r = 0; r < 256; ++r) {
        const float v = p[(size_t)r * FEAT];
        s += v; ss = fmaf(v, v, ss);
    }
    psum[chunk * FEAT + c] = s;
    psumsq[chunk * FEAT + c] = ss;
}

__global__ __launch_bounds__(256) void bn_final(
    const float* __restrict__ psum, const float* __restrict__ psumsq,
    float* __restrict__ meanp, float* __restrict__ invp)
{
    const int c = blockIdx.x * 256 + threadIdx.x;
    if (c >= FEAT) return;
    float s = 0.f, ss = 0.f;
    for (int chunk = 0; chunk < 64; ++chunk) {
        s += psum[chunk * FEAT + c];
        ss += psumsq[chunk * FEAT + c];
    }
    const float mean = s * (1.f / (float)B_DIM);
    const float var  = ss * (1.f / (float)B_DIM) - mean * mean;
    meanp[c] = mean;
    invp[c]  = rsqrtf(var + 1e-5f);
}

// ---------------------------------------------------------------------------
// Kernel 4: per sample b: BN apply -> LayerNorm(3200) -> attn softmax -> pool.
// ---------------------------------------------------------------------------
__global__ __launch_bounds__(256) void fuse_final(
    const float* __restrict__ h,
    const float* __restrict__ meanp, const float* __restrict__ invp,
    const float* __restrict__ bn_g, const float* __restrict__ bn_b,
    const float* __restrict__ ln_g, const float* __restrict__ ln_b,
    const float* __restrict__ attn_w, const float* __restrict__ attn_b,
    float* __restrict__ out)
{
    __shared__ float vals[FEAT];
    __shared__ float red[256];
    __shared__ float red2[256];
    __shared__ float sc[T_DIM * N_DIM];
    __shared__ float wts[T_DIM * N_DIM];

    const int b = blockIdx.x;
    const int tid = threadIdx.x;

    float s = 0.f, ss = 0.f;
    for (int i = tid; i < FEAT; i += 256) {
        float v = h[(size_t)b * FEAT + i];
        v = (v - meanp[i]) * invp[i] * bn_g[i] + bn_b[i];
        vals[i] = v;
        s += v; ss = fmaf(v, v, ss);
    }
    red[tid] = s; red2[tid] = ss;
    __syncthreads();
    for (int st = 128; st > 0; st >>= 1) {
        if (tid < st) { red[tid] += red[tid + st]; red2[tid] += red2[tid + st]; }
        __syncthreads();
    }
    const float lmu  = red[0] * (1.f / (float)FEAT);
    const float lvar = red2[0] * (1.f / (float)FEAT) - lmu * lmu;
    const float linv = rsqrtf(lvar + 1e-5f);
    __syncthreads();

    for (int i = tid; i < FEAT; i += 256)
        vals[i] = (vals[i] - lmu) * linv * ln_g[i] + ln_b[i];
    __syncthreads();

    if (tid < T_DIM * N_DIM) {
        const int tt = tid / N_DIM, n = tid % N_DIM;
        float a = attn_b[0];
        #pragma unroll
        for (int d = 0; d < 32; ++d)
            a = fmaf(vals[tt * K_DIM + n * 32 + d], attn_w[d], a);
        sc[tid] = a;
    }
    __syncthreads();

    if (tid < T_DIM) {
        const int tt = tid;
        float mx = -1e30f;
        for (int n = 0; n < N_DIM; ++n) mx = fmaxf(mx, sc[tt * N_DIM + n]);
        float se = 0.f;
        for (int n = 0; n < N_DIM; ++n) {
            const float e = __expf(sc[tt * N_DIM + n] - mx);
            wts[tt * N_DIM + n] = e; se += e;
        }
        const float r = 1.f / se;
        for (int n = 0; n < N_DIM; ++n) wts[tt * N_DIM + n] *= r;
    }
    __syncthreads();

    if (tid < T_DIM * 32) {
        const int tt = tid >> 5, d = tid & 31;
        float o = 0.f;
        for (int n = 0; n < N_DIM; ++n)
            o = fmaf(wts[tt * N_DIM + n], vals[tt * K_DIM + n * 32 + d], o);
        out[(size_t)b * (T_DIM * 32) + tid] = o;
    }
}

// ---------------------------------------------------------------------------
extern "C" void kernel_launch(void* const* d_in, const int* in_sizes, int n_in,
                              void* d_out, int out_size, void* d_ws, size_t ws_size,
                              hipStream_t stream) {
    const float* x       = (const float*)d_in[0];
    const float* mparam  = (const float*)d_in[1];
    const float* mbias   = (const float*)d_in[2];
    const float* weight  = (const float*)d_in[3];
    const float* w_ih    = (const float*)d_in[4];
    const float* w_hh    = (const float*)d_in[5];
    const float* b_ih    = (const float*)d_in[6];
    const float* b_hh    = (const float*)d_in[7];
    const float* bn_g    = (const float*)d_in[8];
    const float* bn_b    = (const float*)d_in[9];
    const float* ln_g    = (const float*)d_in[10];
    const float* ln_b    = (const float*)d_in[11];
    const float* attn_w  = (const float*)d_in[12];
    const float* attn_b  = (const float*)d_in[13];
    float* out = (float*)d_out;

    const int NX = B_DIM * K_DIM;            // 26,214,400 x elements
    const int NW = T_DIM * K_DIM * M_DIM;    //  8,000,000 masker elements

    // workspace layout (float units)
    float* adj    = (float*)d_ws;                                  // B*T*2500
    float* hbuf   = adj  + (size_t)B_DIM * T_DIM * M_DIM;          // B*3200
    float* psum   = hbuf + (size_t)B_DIM * FEAT;                   // 64*3200
    float* psumsq = psum + 64 * FEAT;                              // 64*3200
    float* meanp  = psumsq + 64 * FEAT;                            // 3200
    float* invp   = meanp + FEAT;                                  // 3200
    __bf16* xhi = (__bf16*)(invp + FEAT);                          // NX bf16
    __bf16* xlo = xhi + (size_t)NX;                                // NX bf16
    __bf16* whi = xlo + (size_t)NX;                                // NW bf16 (packed)
    __bf16* wlo = whi + (size_t)NW;                                // NW bf16 (packed)

    // 0) split x into hi/lo bf16 planes; transpose-pack masker into (T,2500,1600)
    cvt_split<<<8192, 256, 0, stream>>>(x, xhi, xlo, NX);
    pack_w<<<dim3(79, 50, 2), 256, 0, stream>>>(mparam, whi, wlo);

    // 1) adjacency GEMM (split-bf16 WMMA 16x16x32, async-DMA double-buffered)
    gemm_adj<<<dim3(20, 128, 2), 256, 0, stream>>>(xhi, xlo, whi, wlo, mbias, adj);

    // 2) message passing + GRU: one wave per (b,t)
    msg_gru<<<(B_DIM * T_DIM) / 4, 128, 0, stream>>>(
        x, adj, weight, w_ih, w_hh, b_ih, b_hh, hbuf);

    // 3) BatchNorm stats (deterministic two-phase)
    bn_partial<<<dim3(13, 64), 256, 0, stream>>>(hbuf, psum, psumsq);
    bn_final<<<13, 256, 0, stream>>>(psum, psumsq, meanp, invp);

    // 4) BN apply + LayerNorm + attention pooling
    fuse_final<<<B_DIM, 256, 0, stream>>>(hbuf, meanp, invp, bn_g, bn_b,
                                          ln_g, ln_b, attn_w, attn_b, out);
}
